// MoELayer_77558519431579
// MI455X (gfx1250) — matrix operand order repaired and checked by reading, not compiled, not run on hardware
//
#include <hip/hip_runtime.h>
#include <hip/hip_bf16.h>

// Problem constants (from reference): B=2,S=1024 -> T=2048; H=1024; I=512; E=64; K=8; IS=1024
constexpr int TT  = 2048;
constexpr int HH  = 1024;
constexpr int II  = 512;
constexpr int EE  = 64;
constexpr int KK  = 8;
constexpr int ISH = 1024;

typedef __bf16 v16bf __attribute__((ext_vector_type(16)));
typedef float  v8f   __attribute__((ext_vector_type(8)));
typedef int    i4v   __attribute__((vector_size(16)));   // matches builtin's expected pointee

union FragBF { uint4 u4[2]; v16bf v; };

__device__ __forceinline__ unsigned short f2bf(float f) {
  unsigned int u = __float_as_uint(f);
  // round-to-nearest-even truncation to bf16
  return (unsigned short)((u + 0x7FFFu + ((u >> 16) & 1u)) >> 16);
}

__device__ __forceinline__ float fast_rcp(float v) {
#if __has_builtin(__builtin_amdgcn_rcpf)
  return __builtin_amdgcn_rcpf(v);
#else
  return 1.0f / v;
#endif
}
__device__ __forceinline__ float silu_f(float v) {
  return v * fast_rcp(1.0f + __expf(-v));   // v_rcp_f32 instead of full fdiv chain
}

// ---- gfx1250 async global->LDS copy (ASYNCcnt path) ------------------------
__device__ __forceinline__ void async_copy_b128(const void* gsrc, void* ldst) {
#if __has_builtin(__builtin_amdgcn_global_load_async_to_lds_b128)
  __builtin_amdgcn_global_load_async_to_lds_b128(
      (i4v __attribute__((address_space(1)))*)(void*)gsrc,
      (i4v __attribute__((address_space(3)))*)ldst, 0, 0);
#else
  unsigned lofs = (unsigned)(unsigned long long)
      (__attribute__((address_space(3))) void*)ldst;
  asm volatile("global_load_async_to_lds_b128 %0, %1, off"
               :: "v"(lofs), "v"(gsrc) : "memory");
#endif
}
__device__ __forceinline__ void wait_async0() {
#if __has_builtin(__builtin_amdgcn_s_wait_asynccnt)
  __builtin_amdgcn_s_wait_asynccnt(0);
#else
  asm volatile("s_wait_asynccnt 0x0" ::: "memory");
#endif
}

// ---------------------------------------------------------------------------
// Kernel 1: gate logits + softmax + top-8 + renorm; emit idx, per-token routing,
// and per-expert counts.
// ---------------------------------------------------------------------------
__global__ __launch_bounds__(64)
void k_gate(const float* __restrict__ x, const float* __restrict__ gw,
            int* __restrict__ counts, int* __restrict__ tk_e,
            float* __restrict__ tk_w, int* __restrict__ idx_out) {
  __shared__ float sx[HH];
  __shared__ float sl[EE];
  const int t = blockIdx.x;
  const int e = threadIdx.x;            // 0..63 (one expert per thread)
  const float4* xr  = (const float4*)(x + (size_t)t * HH);
  float4*       sx4 = (float4*)sx;
  #pragma unroll
  for (int j = 0; j < 4; ++j) sx4[e * 4 + j] = xr[e * 4 + j];
  __syncthreads();
  const float4* g4 = (const float4*)(gw + (size_t)e * HH);
  float acc = 0.f;
  for (int j = 0; j < HH / 4; ++j) {
    float4 a = sx4[j], b = g4[j];
    acc += a.x * b.x + a.y * b.y + a.z * b.z + a.w * b.w;
  }
  sl[e] = acc;
  __syncthreads();
  if (e == 0) {
    float m = sl[0];
    #pragma unroll 1
    for (int i = 1; i < EE; ++i) m = fmaxf(m, sl[i]);
    float sc[EE];
    float s = 0.f;
    #pragma unroll 1
    for (int i = 0; i < EE; ++i) { sc[i] = __expf(sl[i] - m); s += sc[i]; }
    const float inv = 1.f / s;
    #pragma unroll 1
    for (int i = 0; i < EE; ++i) sc[i] *= inv;
    int sel[KK]; float w[KK]; float wsum = 0.f;
    #pragma unroll 1
    for (int k = 0; k < KK; ++k) {
      int bi = 0; float bv = -1.f;
      for (int i = 0; i < EE; ++i) if (sc[i] > bv) { bv = sc[i]; bi = i; }
      sel[k] = bi; w[k] = bv; sc[bi] = -2.f; wsum += bv;
    }
    const float rinv = 1.f / (wsum + 1e-20f);
    #pragma unroll 1
    for (int k = 0; k < KK; ++k) {
      tk_e[t * KK + k]    = sel[k];
      tk_w[t * KK + k]    = w[k] * rinv;
      idx_out[t * KK + k] = sel[k];
      atomicAdd(&counts[sel[k]], 1);
    }
  }
}

// ---------------------------------------------------------------------------
// Kernel 2: exclusive prefix sum of counts -> offsets; zero cursors.
// ---------------------------------------------------------------------------
__global__ void k_scan(const int* __restrict__ counts, int* __restrict__ offsets,
                       int* __restrict__ cursor) {
  if (threadIdx.x == 0) {
    int run = 0;
    for (int e2 = 0; e2 < EE; ++e2) { offsets[e2] = run; run += counts[e2]; }
  }
  if (threadIdx.x < EE) cursor[threadIdx.x] = 0;
}

// ---------------------------------------------------------------------------
// Kernel 3: build per-expert compact token/weight lists.
// ---------------------------------------------------------------------------
__global__ void k_scatter(const int* __restrict__ tk_e, const float* __restrict__ tk_w,
                          const int* __restrict__ offsets, int* __restrict__ cursor,
                          int* __restrict__ tok_list, float* __restrict__ w_list) {
  int i = blockIdx.x * blockDim.x + threadIdx.x;
  if (i >= TT * KK) return;
  int e2  = tk_e[i];
  int pos = atomicAdd(&cursor[e2], 1);
  tok_list[offsets[e2] + pos] = i / KK;
  w_list[offsets[e2] + pos]   = tk_w[i];
}

// ---------------------------------------------------------------------------
// Kernel 4: shared expert up: act = bf16( silu(x@swg) * (x@swu) )   [T x IS]
// 64x64 tile per block, K-step 32, 4 waves, wmma bf16.
// ---------------------------------------------------------------------------
__global__ __launch_bounds__(128)
void k_shared_up(const float* __restrict__ x, const float* __restrict__ swg,
                 const float* __restrict__ swu, unsigned short* __restrict__ act) {
  __shared__ __align__(16) unsigned short sA[64][32];
  __shared__ __align__(16) unsigned short sBg[64][32];
  __shared__ __align__(16) unsigned short sBu[64][32];
  const int tid = threadIdx.x, lane = tid & 31, wave = tid >> 5;
  const int t0 = blockIdx.x * 64, n0 = blockIdx.y * 64;
  v8f accg[4], accu[4];
  #pragma unroll
  for (int i = 0; i < 4; ++i) { accg[i] = {}; accu[i] = {}; }
  const int arow = tid >> 1, akoff = (tid & 1) * 16;
  const int brow = tid >> 2, bnoff = (tid & 3) * 16;
  const int rowA = wave * 16 + (lane & 15);
  const int kbA  = 16 * (lane >> 4);
  const int kbB  = 16 * (lane >> 4);
  for (int kk = 0; kk < HH / 32; ++kk) {
    const int k0 = kk * 32;
    const float4* ap = (const float4*)(x   + (size_t)(t0 + arow) * HH  + k0 + akoff);
    const float4* gp = (const float4*)(swg + (size_t)(k0 + brow) * ISH + n0 + bnoff);
    const float4* up = (const float4*)(swu + (size_t)(k0 + brow) * ISH + n0 + bnoff);
    __builtin_prefetch((const void*)(ap + 8), 0, 1);
    unsigned short* sa = &sA[arow][akoff];
    #pragma unroll
    for (int j = 0; j < 4; ++j) {
      float4 v = ap[j];
      sa[4*j+0] = f2bf(v.x); sa[4*j+1] = f2bf(v.y); sa[4*j+2] = f2bf(v.z); sa[4*j+3] = f2bf(v.w);
    }
    #pragma unroll
    for (int j = 0; j < 4; ++j) {
      float4 vg = gp[j]; float4 vu = up[j];
      int nb = bnoff + 4 * j;
      sBg[nb+0][brow] = f2bf(vg.x); sBg[nb+1][brow] = f2bf(vg.y);
      sBg[nb+2][brow] = f2bf(vg.z); sBg[nb+3][brow] = f2bf(vg.w);
      sBu[nb+0][brow] = f2bf(vu.x); sBu[nb+1][brow] = f2bf(vu.y);
      sBu[nb+2][brow] = f2bf(vu.z); sBu[nb+3][brow] = f2bf(vu.w);
    }
    __syncthreads();
    FragBF fa;
    fa.u4[0] = *(const uint4*)&sA[rowA][kbA];
    fa.u4[1] = *(const uint4*)&sA[rowA][kbA + 8];
    #pragma unroll
    for (int ns = 0; ns < 4; ++ns) {
      const int col = ns * 16 + (lane & 15);
      FragBF fbg, fbu;
      fbg.u4[0] = *(const uint4*)&sBg[col][kbB];
      fbg.u4[1] = *(const uint4*)&sBg[col][kbB + 8];
      fbu.u4[0] = *(const uint4*)&sBu[col][kbB];
      fbu.u4[1] = *(const uint4*)&sBu[col][kbB + 8];
      accg[ns] = __builtin_amdgcn_wmma_f32_16x16x32_bf16(false, fa.v, false, fbg.v,
                                                         (short)0, accg[ns], false, false);
      accu[ns] = __builtin_amdgcn_wmma_f32_16x16x32_bf16(false, fa.v, false, fbu.v,
                                                         (short)0, accu[ns], false, false);
    }
    __syncthreads();
  }
  const int mofs = 8 * (lane >> 4);
  #pragma unroll
  for (int ns = 0; ns < 4; ++ns)
    #pragma unroll
    for (int v = 0; v < 8; ++v) {
      int r = t0 + wave * 16 + v + mofs;
      int c = n0 + ns * 16 + (lane & 15);
      act[(size_t)r * ISH + c] = f2bf(silu_f(accg[ns][v]) * accu[ns][v]);
    }
}

// ---------------------------------------------------------------------------
// Kernel 5: shared expert down: out = act @ swd   (plain stores; initializes out)
// A-tile (bf16) streamed with async global->LDS copies.
// ---------------------------------------------------------------------------
__global__ __launch_bounds__(128)
void k_shared_down(const unsigned short* __restrict__ act, const float* __restrict__ swd,
                   float* __restrict__ out) {
  __shared__ __align__(16) unsigned short sA[64][32];
  __shared__ __align__(16) unsigned short sB[64][32];
  const int tid = threadIdx.x, lane = tid & 31, wave = tid >> 5;
  const int t0 = blockIdx.x * 64, n0 = blockIdx.y * 64;
  v8f acc[4];
  #pragma unroll
  for (int i = 0; i < 4; ++i) acc[i] = {};
  const int arow = tid >> 1, akoff = (tid & 1) * 16;
  const int brow = tid >> 2, bnoff = (tid & 3) * 16;
  const int rowA = wave * 16 + (lane & 15);
  const int kbA  = 16 * (lane >> 4);
  const int kbB  = 16 * (lane >> 4);
  for (int kk = 0; kk < ISH / 32; ++kk) {
    const int k0 = kk * 32;
    const unsigned short* agp = act + (size_t)(t0 + arow) * ISH + k0 + akoff;
    // async global->LDS: 16 bf16 (2 x b128) per thread, no VGPR round-trip
    async_copy_b128(agp,     &sA[arow][akoff]);
    async_copy_b128(agp + 8, &sA[arow][akoff + 8]);
    const float4* bp = (const float4*)(swd + (size_t)(k0 + brow) * HH + n0 + bnoff);
    __builtin_prefetch((const void*)(bp + 8), 0, 1);
    #pragma unroll
    for (int j = 0; j < 4; ++j) {
      float4 v = bp[j];
      int nb = bnoff + 4 * j;
      sB[nb+0][brow] = f2bf(v.x); sB[nb+1][brow] = f2bf(v.y);
      sB[nb+2][brow] = f2bf(v.z); sB[nb+3][brow] = f2bf(v.w);
    }
    wait_async0();
    __syncthreads();
    FragBF fa;
    fa.u4[0] = *(const uint4*)&sA[rowA][kbA];
    fa.u4[1] = *(const uint4*)&sA[rowA][kbA + 8];
    #pragma unroll
    for (int ns = 0; ns < 4; ++ns) {
      const int col = ns * 16 + (lane & 15);
      FragBF fb;
      fb.u4[0] = *(const uint4*)&sB[col][kbB];
      fb.u4[1] = *(const uint4*)&sB[col][kbB + 8];
      acc[ns] = __builtin_amdgcn_wmma_f32_16x16x32_bf16(false, fa.v, false, fb.v,
                                                        (short)0, acc[ns], false, false);
    }
    __syncthreads();
  }
  const int mofs = 8 * (lane >> 4);
  #pragma unroll
  for (int ns = 0; ns < 4; ++ns)
    #pragma unroll
    for (int v = 0; v < 8; ++v) {
      int r = t0 + wave * 16 + v + mofs;
      int c = n0 + ns * 16 + (lane & 15);
      out[(size_t)r * HH + c] = acc[ns][v];
    }
}

// ---------------------------------------------------------------------------
// Kernel 6: routed up per expert (gathered rows):
//   act_rt[slot] = bf16( silu(Xe@wg[e]) * (Xe@wu[e]) )   [Ne x I]
// ---------------------------------------------------------------------------
__global__ __launch_bounds__(128)
void k_routed_up(const float* __restrict__ x, const float* __restrict__ wg,
                 const float* __restrict__ wu, const int* __restrict__ counts,
                 const int* __restrict__ offsets, const int* __restrict__ tok_list,
                 unsigned short* __restrict__ act) {
  const int e2 = blockIdx.z;
  const int Ne = counts[e2];
  const int m0 = blockIdx.x * 64;
  if (m0 >= Ne) return;
  const int off = offsets[e2];
  const int n0  = blockIdx.y * 64;
  const float* wgp = wg + (size_t)e2 * HH * II;
  const float* wup = wu + (size_t)e2 * HH * II;
  __shared__ __align__(16) unsigned short sA[64][32];
  __shared__ __align__(16) unsigned short sBg[64][32];
  __shared__ __align__(16) unsigned short sBu[64][32];
  const int tid = threadIdx.x, lane = tid & 31, wave = tid >> 5;
  v8f accg[4], accu[4];
  #pragma unroll
  for (int i = 0; i < 4; ++i) { accg[i] = {}; accu[i] = {}; }
  const int arow = tid >> 1, akoff = (tid & 1) * 16;
  const int brow = tid >> 2, bnoff = (tid & 3) * 16;
  const int rgather = m0 + arow;
  const int tok = (rgather < Ne) ? tok_list[off + rgather] : tok_list[off];
  const int rowA = wave * 16 + (lane & 15);
  const int kbA  = 16 * (lane >> 4);
  const int kbB  = 16 * (lane >> 4);
  for (int kk = 0; kk < HH / 32; ++kk) {
    const int k0 = kk * 32;
    const float4* ap = (const float4*)(x   + (size_t)tok * HH + k0 + akoff);
    const float4* gp = (const float4*)(wgp + (size_t)(k0 + brow) * II + n0 + bnoff);
    const float4* up = (const float4*)(wup + (size_t)(k0 + brow) * II + n0 + bnoff);
    __builtin_prefetch((const void*)(gp + 8), 0, 1);
    unsigned short* sa = &sA[arow][akoff];
    #pragma unroll
    for (int j = 0; j < 4; ++j) {
      float4 v = ap[j];
      sa[4*j+0] = f2bf(v.x); sa[4*j+1] = f2bf(v.y); sa[4*j+2] = f2bf(v.z); sa[4*j+3] = f2bf(v.w);
    }
    #pragma unroll
    for (int j = 0; j < 4; ++j) {
      float4 vg = gp[j]; float4 vu = up[j];
      int nb = bnoff + 4 * j;
      sBg[nb+0][brow] = f2bf(vg.x); sBg[nb+1][brow] = f2bf(vg.y);
      sBg[nb+2][brow] = f2bf(vg.z); sBg[nb+3][brow] = f2bf(vg.w);
      sBu[nb+0][brow] = f2bf(vu.x); sBu[nb+1][brow] = f2bf(vu.y);
      sBu[nb+2][brow] = f2bf(vu.z); sBu[nb+3][brow] = f2bf(vu.w);
    }
    __syncthreads();
    FragBF fa;
    fa.u4[0] = *(const uint4*)&sA[rowA][kbA];
    fa.u4[1] = *(const uint4*)&sA[rowA][kbA + 8];
    #pragma unroll
    for (int ns = 0; ns < 4; ++ns) {
      const int col = ns * 16 + (lane & 15);
      FragBF fbg, fbu;
      fbg.u4[0] = *(const uint4*)&sBg[col][kbB];
      fbg.u4[1] = *(const uint4*)&sBg[col][kbB + 8];
      fbu.u4[0] = *(const uint4*)&sBu[col][kbB];
      fbu.u4[1] = *(const uint4*)&sBu[col][kbB + 8];
      accg[ns] = __builtin_amdgcn_wmma_f32_16x16x32_bf16(false, fa.v, false, fbg.v,
                                                         (short)0, accg[ns], false, false);
      accu[ns] = __builtin_amdgcn_wmma_f32_16x16x32_bf16(false, fa.v, false, fbu.v,
                                                         (short)0, accu[ns], false, false);
    }
    __syncthreads();
  }
  const int mofs = 8 * (lane >> 4);
  #pragma unroll
  for (int ns = 0; ns < 4; ++ns)
    #pragma unroll
    for (int v = 0; v < 8; ++v) {
      int rloc = m0 + wave * 16 + v + mofs;
      if (rloc < Ne) {
        int c = n0 + ns * 16 + (lane & 15);
        act[(size_t)(off + rloc) * II + c] = f2bf(silu_f(accg[ns][v]) * accu[ns][v]);
      }
    }
}

// ---------------------------------------------------------------------------
// Kernel 7: routed down per expert: out[tok] += w * (act_e @ wd[e])  (atomic f32)
// A-tile (bf16) streamed with async global->LDS copies for in-range rows.
// ---------------------------------------------------------------------------
__global__ __launch_bounds__(128)
void k_routed_down(const unsigned short* __restrict__ act, const float* __restrict__ wd,
                   const int* __restrict__ counts, const int* __restrict__ offsets,
                   const int* __restrict__ tok_list, const float* __restrict__ w_list,
                   float* __restrict__ out) {
  const int e2 = blockIdx.z;
  const int Ne = counts[e2];
  const int m0 = blockIdx.x * 64;
  if (m0 >= Ne) return;
  const int off = offsets[e2];
  const int n0  = blockIdx.y * 64;
  const float* wdp = wd + (size_t)e2 * II * HH;
  __shared__ __align__(16) unsigned short sA[64][32];
  __shared__ __align__(16) unsigned short sB[64][32];
  const int tid = threadIdx.x, lane = tid & 31, wave = tid >> 5;
  v8f acc[4];
  #pragma unroll
  for (int i = 0; i < 4; ++i) acc[i] = {};
  const int arow = tid >> 1, akoff = (tid & 1) * 16;
  const int brow = tid >> 2, bnoff = (tid & 3) * 16;
  const bool arow_ok = (m0 + arow) < Ne;
  const int rowA = wave * 16 + (lane & 15);
  const int kbA  = 16 * (lane >> 4);
  const int kbB  = 16 * (lane >> 4);
  for (int kk = 0; kk < II / 32; ++kk) {
    const int k0 = kk * 32;
    if (arow_ok) {
      const unsigned short* agp = act + (size_t)(off + m0 + arow) * II + k0 + akoff;
      async_copy_b128(agp,     &sA[arow][akoff]);
      async_copy_b128(agp + 8, &sA[arow][akoff + 8]);
    } else {
      uint4 z = make_uint4(0u, 0u, 0u, 0u);
      uint4* sa4 = (uint4*)&sA[arow][akoff];
      sa4[0] = z; sa4[1] = z;
    }
    const float4* bp = (const float4*)(wdp + (size_t)(k0 + brow) * HH + n0 + bnoff);
    __builtin_prefetch((const void*)(bp + 8), 0, 1);
    #pragma unroll
    for (int j = 0; j < 4; ++j) {
      float4 v = bp[j];
      int nb = bnoff + 4 * j;
      sB[nb+0][brow] = f2bf(v.x); sB[nb+1][brow] = f2bf(v.y);
      sB[nb+2][brow] = f2bf(v.z); sB[nb+3][brow] = f2bf(v.w);
    }
    wait_async0();
    __syncthreads();
    FragBF fa;
    fa.u4[0] = *(const uint4*)&sA[rowA][kbA];
    fa.u4[1] = *(const uint4*)&sA[rowA][kbA + 8];
    #pragma unroll
    for (int ns = 0; ns < 4; ++ns) {
      const int col = ns * 16 + (lane & 15);
      FragBF fb;
      fb.u4[0] = *(const uint4*)&sB[col][kbB];
      fb.u4[1] = *(const uint4*)&sB[col][kbB + 8];
      acc[ns] = __builtin_amdgcn_wmma_f32_16x16x32_bf16(false, fa.v, false, fb.v,
                                                        (short)0, acc[ns], false, false);
    }
    __syncthreads();
  }
  const int mofs = 8 * (lane >> 4);
  #pragma unroll
  for (int ns = 0; ns < 4; ++ns)
    #pragma unroll
    for (int v = 0; v < 8; ++v) {
      int rloc = m0 + wave * 16 + v + mofs;
      if (rloc < Ne) {
        int t = tok_list[off + rloc];
        float w = w_list[off + rloc];
        int c = n0 + ns * 16 + (lane & 15);
        atomicAdd(&out[(size_t)t * HH + c], w * acc[ns][v]);
      }
    }
}

// ---------------------------------------------------------------------------
extern "C" void kernel_launch(void* const* d_in, const int* in_sizes, int n_in,
                              void* d_out, int out_size, void* d_ws, size_t ws_size,
                              hipStream_t stream) {
  const float* x      = (const float*)d_in[0];
  const float* gate_w = (const float*)d_in[1];
  const float* wg     = (const float*)d_in[2];
  const float* wu     = (const float*)d_in[3];
  const float* wd     = (const float*)d_in[4];
  const float* swg    = (const float*)d_in[5];
  const float* swu    = (const float*)d_in[6];
  const float* swd    = (const float*)d_in[7];

  float* out     = (float*)d_out;
  int*   idx_out = (int*)d_out + (size_t)TT * HH;   // topk_idx appended after [B,S,H] floats

  char* ws = (char*)d_ws;
  int*   counts   = (int*)(ws);
  int*   offsets  = (int*)(ws + 256);
  int*   cursor   = (int*)(ws + 512);
  int*   tk_e     = (int*)(ws + 1024);
  float* tk_w     = (float*)(ws + 1024 + 1 * 65536);
  int*   tok_list = (int*)(ws + 1024 + 2 * 65536);
  float* w_list   = (float*)(ws + 1024 + 3 * 65536);
  unsigned short* act_sh = (unsigned short*)(ws + 1024 + 4 * 65536);
  unsigned short* act_rt = (unsigned short*)((char*)act_sh + (size_t)TT * ISH * 2);

  (void)hipMemsetAsync(ws, 0, 768, stream);  // counts + offsets + cursor

  k_gate<<<TT, 64, 0, stream>>>(x, gate_w, counts, tk_e, tk_w, idx_out);
  k_scan<<<1, 64, 0, stream>>>(counts, offsets, cursor);
  k_scatter<<<(TT * KK + 255) / 256, 256, 0, stream>>>(tk_e, tk_w, offsets, cursor,
                                                       tok_list, w_list);
  k_shared_up<<<dim3(TT / 64, ISH / 64), 128, 0, stream>>>(x, swg, swu, act_sh);
  k_shared_down<<<dim3(TT / 64, HH / 64), 128, 0, stream>>>(act_sh, swd, out);
  k_routed_up<<<dim3(TT / 64, II / 64, EE), 128, 0, stream>>>(x, wg, wu, counts, offsets,
                                                              tok_list, act_rt);
  k_routed_down<<<dim3(TT / 64, HH / 64, EE), 128, 0, stream>>>(act_rt, wd, counts, offsets,
                                                                tok_list, w_list, out);
}
